// SSIDDIBlock_7327214207390
// MI455X (gfx1250) — matrix operand order repaired and checked by reading, not compiled, not run on hardware
//
#include <hip/hip_runtime.h>
#include <hip/hip_bf16.h>

typedef __attribute__((ext_vector_type(16))) __bf16 v16bf;
typedef __attribute__((ext_vector_type(8)))  float  v8f;

#define HEADS 2
#define CH 64
#define INCH 64
#define NEG_SLOPE 0.2f

// ---- order-preserving float <-> int mapping for atomic max on floats ----
__device__ __forceinline__ int f2oi(float f) {
  int i = __float_as_int(f);
  return (i >= 0) ? i : (i ^ 0x7FFFFFFF);
}
__device__ __forceinline__ float oi2f(int i) {
  return __int_as_float((i >= 0) ? i : (i ^ 0x7FFFFFFF));
}

// ---------------- init helpers ----------------
__global__ void k_zero_f32(float* __restrict__ p, int n) {
  int t = blockIdx.x * blockDim.x + threadIdx.x;
  if (t < n) p[t] = 0.0f;
}
__global__ void k_fill_i32(int* __restrict__ p, int n, int v) {
  int t = blockIdx.x * blockDim.x + threadIdx.x;
  if (t < n) p[t] = v;
}

// ---------------- projection GEMM: xp[N,128] = x[N,64] @ W[128,64]^T ----------------
// One wave per 16x16 output tile; blockDim=256 (8 waves) covers all 128 output cols.
// bf16 operands (b128 loads + packed converts), fp32 accumulation in v_wmma_f32_16x16x32_bf16.
__global__ void k_proj_wmma(const float* __restrict__ x, const float* __restrict__ W,
                            float* __restrict__ xp, int N) {
  const int lane = threadIdx.x & 31;
  const int wave = threadIdx.x >> 5;     // 0..7 -> output column tile
  const int half = lane >> 4;            // 0 / 1 (lane group)
  const int m    = lane & 15;
  const int row0 = blockIdx.x * 16;
  const int j0   = wave * 16;

  const int arow = (row0 + m < N) ? (row0 + m) : (N - 1);

  v8f c = {};
#pragma unroll
  for (int ks = 0; ks < INCH; ks += 32) {
    // A (16x32, MxK): lanes 0-15 hold M rows with K {ks..ks+7, ks+16..ks+23};
    // lanes 16-31 hold K {ks+8..ks+15, ks+24..ks+31}. All fragments 16B aligned.
    const float4* xa = reinterpret_cast<const float4*>(x + (size_t)arow * INCH + ks + half * 8);
    float4 p0 = xa[0];
    float4 p1 = xa[1];
    const float4* xb = reinterpret_cast<const float4*>(x + (size_t)arow * INCH + ks + half * 8 + 16);
    float4 q0 = xb[0];
    float4 q1 = xb[1];
    // B (32x16, KxN), B[k][n] = W[j0+n][k]: same K split as A,
    // lanes 0-15 -> K {ks..ks+7, ks+16..ks+23}, lanes 16-31 -> K {ks+8..ks+15, ks+24..ks+31}.
    const float4* wr = reinterpret_cast<const float4*>(W + (size_t)(j0 + m) * INCH + ks + half * 8);
    float4 w0 = wr[0], w1 = wr[1], w2 = wr[4], w3 = wr[5];

    v16bf a, b;
    a[0]  = (__bf16)p0.x; a[1]  = (__bf16)p0.y; a[2]  = (__bf16)p0.z; a[3]  = (__bf16)p0.w;
    a[4]  = (__bf16)p1.x; a[5]  = (__bf16)p1.y; a[6]  = (__bf16)p1.z; a[7]  = (__bf16)p1.w;
    a[8]  = (__bf16)q0.x; a[9]  = (__bf16)q0.y; a[10] = (__bf16)q0.z; a[11] = (__bf16)q0.w;
    a[12] = (__bf16)q1.x; a[13] = (__bf16)q1.y; a[14] = (__bf16)q1.z; a[15] = (__bf16)q1.w;

    b[0]  = (__bf16)w0.x; b[1]  = (__bf16)w0.y; b[2]  = (__bf16)w0.z; b[3]  = (__bf16)w0.w;
    b[4]  = (__bf16)w1.x; b[5]  = (__bf16)w1.y; b[6]  = (__bf16)w1.z; b[7]  = (__bf16)w1.w;
    b[8]  = (__bf16)w2.x; b[9]  = (__bf16)w2.y; b[10] = (__bf16)w2.z; b[11] = (__bf16)w2.w;
    b[12] = (__bf16)w3.x; b[13] = (__bf16)w3.y; b[14] = (__bf16)w3.z; b[15] = (__bf16)w3.w;

    c = __builtin_amdgcn_wmma_f32_16x16x32_bf16(
        /*neg_a=*/false, a, /*neg_b=*/false, b,
        /*c_mod=*/(short)0, c, /*reuse_a=*/false, /*reuse_b=*/false);
  }

  // C/D layout: VGPR r -> M = r + 8*half, N = m. Full-tile fast path (wave-uniform branch).
  float* out0 = xp + (size_t)(row0 + half * 8) * (HEADS * CH) + j0 + m;
  if (row0 + 16 <= N) {
#pragma unroll
    for (int r = 0; r < 8; ++r)
      out0[(size_t)r * (HEADS * CH)] = c[r];
  } else {
#pragma unroll
    for (int r = 0; r < 8; ++r) {
      int row = row0 + r + half * 8;
      if (row < N) xp[(size_t)row * (HEADS * CH) + j0 + m] = c[r];
    }
  }
}

// ---------------- per-node attention coefficients ----------------
__global__ void k_att(const float* __restrict__ xp, const float* __restrict__ att_s,
                      const float* __restrict__ att_d, float* __restrict__ a_src,
                      float* __restrict__ a_dst, int N) {
  int t = blockIdx.x * blockDim.x + threadIdx.x;   // over N*HEADS
  if (t >= N * HEADS) return;
  int n = t >> 1, h = t & 1;
  const float* row = xp + (size_t)n * (HEADS * CH) + h * CH;
  const float* as  = att_s + h * CH;
  const float* ad  = att_d + h * CH;
  float ss = 0.0f, sd = 0.0f;
#pragma unroll 8
  for (int c = 0; c < CH; ++c) {
    float v = row[c];
    ss += v * as[c];
    sd += v * ad[c];
  }
  a_src[t] = ss;
  a_dst[t] = sd;
}

// ---------------- segment max over destination (softmax pass 1) ----------------
__global__ void k_edge_max(const int* __restrict__ ei, int E, int ET,
                           const float* __restrict__ a_src, const float* __restrict__ a_dst,
                           int* __restrict__ m_int) {
  int e = blockIdx.x * blockDim.x + threadIdx.x;
  if (e >= ET) return;
  int s = (e < E) ? ei[e]     : (e - E);   // self-loop for e >= E
  int d = (e < E) ? ei[E + e] : (e - E);
#pragma unroll
  for (int h = 0; h < HEADS; ++h) {
    float v = a_src[s * HEADS + h] + a_dst[d * HEADS + h];
    v = (v > 0.0f) ? v : NEG_SLOPE * v;
    atomicMax(&m_int[d * HEADS + h], f2oi(v));
  }
}

// ---------------- exp + segment sum (softmax pass 2), stores exp numerators ----------------
__global__ void k_edge_expsum(const int* __restrict__ ei, int E, int ET,
                              const float* __restrict__ a_src, const float* __restrict__ a_dst,
                              const int* __restrict__ m_int, float* __restrict__ p_num,
                              float* __restrict__ s_sum) {
  int e = blockIdx.x * blockDim.x + threadIdx.x;
  if (e >= ET) return;
  int s = (e < E) ? ei[e]     : (e - E);
  int d = (e < E) ? ei[E + e] : (e - E);
#pragma unroll
  for (int h = 0; h < HEADS; ++h) {
    float v = a_src[s * HEADS + h] + a_dst[d * HEADS + h];
    v = (v > 0.0f) ? v : NEG_SLOPE * v;
    float p = __expf(v - oi2f(m_int[d * HEADS + h]));
    p_num[e * HEADS + h] = p;
    atomicAdd(&s_sum[d * HEADS + h], p);
  }
}

// ---------------- weighted message scatter with fused head-mean ----------------
// One WAVE per edge, 2 channels per lane (float2 gathers):
//   h_nodes[d,c] += 0.5*(a0*xp[s,c] + a1*xp[s,64+c])
__global__ void k_scatter(const int* __restrict__ ei, int E, int ET,
                          const float* __restrict__ xp,
                          const float* __restrict__ p_num, const float* __restrict__ s_sum,
                          float* __restrict__ out_nodes) {
  long long t = (long long)blockIdx.x * blockDim.x + threadIdx.x;
  int e    = (int)(t >> 5);
  int lane = (int)(t & 31);
  if (e >= ET) return;
  int s = (e < E) ? ei[e]     : (e - E);
  int d = (e < E) ? ei[E + e] : (e - E);
  float a0 = p_num[e * 2 + 0] / s_sum[d * 2 + 0];
  float a1 = p_num[e * 2 + 1] / s_sum[d * 2 + 1];
  const float2* row = reinterpret_cast<const float2*>(xp + (size_t)s * (HEADS * CH));
  float2 h0 = row[lane];        // head 0, cols 2*lane, 2*lane+1
  float2 h1 = row[32 + lane];   // head 1, same cols
  float v0 = 0.5f * (a0 * h0.x + a1 * h1.x);
  float v1 = 0.5f * (a0 * h0.y + a1 * h1.y);
  float* o = out_nodes + (size_t)d * CH + 2 * lane;
  atomicAdd(o, v0);
  atomicAdd(o + 1, v1);
}

// ---------------- bias + graph pooling accumulation ----------------
__global__ void k_finalize(float* __restrict__ out_nodes, const float* __restrict__ bias,
                           const int* __restrict__ batch, float* __restrict__ gsum,
                           int* __restrict__ gcnt, int N) {
  int t = blockIdx.x * blockDim.x + threadIdx.x;   // over N*CH
  if (t >= N * CH) return;
  int n = t >> 6, c = t & 63;
  float h = out_nodes[t] + bias[c];
  out_nodes[t] = h;
  int g = batch[n];
  atomicAdd(&gsum[g * CH + c], h);
  if (c == 0) atomicAdd(&gcnt[g], 1);
}

__global__ void k_pool(const float* __restrict__ gsum, const int* __restrict__ gcnt,
                       float* __restrict__ out_graphs, int G) {
  int t = blockIdx.x * blockDim.x + threadIdx.x;   // over G*CH
  if (t >= G * CH) return;
  int g = t >> 6;
  float cnt = (float)gcnt[g];
  out_graphs[t] = gsum[t] / fmaxf(cnt, 1.0f);
}

extern "C" void kernel_launch(void* const* d_in, const int* in_sizes, int n_in,
                              void* d_out, int out_size, void* d_ws, size_t ws_size,
                              hipStream_t stream) {
  const float* x       = (const float*)d_in[0];
  const int*   ei      = (const int*)  d_in[1];
  const int*   batch   = (const int*)  d_in[2];
  const float* W       = (const float*)d_in[3];
  const float* att_src = (const float*)d_in[4];
  const float* att_dst = (const float*)d_in[5];
  const float* bias    = (const float*)d_in[6];

  const int N  = in_sizes[0] / INCH;
  const int E  = in_sizes[1] / 2;
  const int G  = (out_size - N * CH) / CH;
  const int ET = E + N;   // edges + self loops

  // workspace layout (floats unless noted)
  float* xp    = (float*)d_ws;                         // N * 128
  float* a_src = xp    + (size_t)N * (HEADS * CH);     // N * 2
  float* a_dst = a_src + (size_t)N * HEADS;            // N * 2
  int*   m_int = (int*)(a_dst + (size_t)N * HEADS);    // N * 2 (ordered-int max)
  float* s_sum = (float*)(m_int + (size_t)N * HEADS);  // N * 2
  float* p_num = s_sum + (size_t)N * HEADS;            // ET * 2
  float* gsum  = p_num + (size_t)ET * HEADS;           // G * 64
  int*   gcnt  = (int*)(gsum + (size_t)G * CH);        // G

  float* out_nodes  = (float*)d_out;                   // N * 64
  float* out_graphs = out_nodes + (size_t)N * CH;      // G * 64

  const int B = 256;

  // init accumulators (every call -> deterministic under graph replay)
  k_zero_f32<<<(N * CH + B - 1) / B, B, 0, stream>>>(out_nodes, N * CH);
  k_zero_f32<<<(N * HEADS + B - 1) / B, B, 0, stream>>>(s_sum, N * HEADS);
  k_zero_f32<<<(G * CH + B - 1) / B, B, 0, stream>>>(gsum, G * CH);
  k_fill_i32<<<(N * HEADS + B - 1) / B, B, 0, stream>>>(m_int, N * HEADS, (int)0x80000000);
  k_fill_i32<<<(G + B - 1) / B, B, 0, stream>>>(gcnt, G, 0);

  // 1) projection via WMMA (8 waves/block cover full 128-wide row tile)
  k_proj_wmma<<<(N + 15) / 16, 256, 0, stream>>>(x, W, xp, N);

  // 2) attention coefficients
  k_att<<<(N * HEADS + B - 1) / B, B, 0, stream>>>(xp, att_src, att_dst, a_src, a_dst, N);

  // 3) segment softmax over incoming edges
  k_edge_max<<<(ET + B - 1) / B, B, 0, stream>>>(ei, E, ET, a_src, a_dst, m_int);
  k_edge_expsum<<<(ET + B - 1) / B, B, 0, stream>>>(ei, E, ET, a_src, a_dst, m_int, p_num, s_sum);

  // 4) weighted scatter with fused head mean (one wave per edge, float2 per lane)
  long long tot = (long long)ET * 32;
  k_scatter<<<(int)((tot + B - 1) / B), B, 0, stream>>>(ei, E, ET, xp, p_num, s_sum, out_nodes);

  // 5) bias + pooling
  k_finalize<<<(N * CH + B - 1) / B, B, 0, stream>>>(out_nodes, bias, batch, gsum, gcnt, N);
  k_pool<<<(G * CH + B - 1) / B, B, 0, stream>>>(gsum, gcnt, out_graphs, G);
}
